// LTE_84945863180636
// MI455X (gfx1250) — matrix operand order repaired and checked by reading, not compiled, not run on hardware
//
#include <hip/hip_runtime.h>
#include <cmath>

typedef __attribute__((ext_vector_type(16))) _Float16 v16h;
typedef __attribute__((ext_vector_type(8)))  _Float16 v8h;
typedef __attribute__((ext_vector_type(8)))  float    v8f;

#define Bn   2
#define Hh   64
#define Ww   64
#define Qq   65536
#define HID  256
#define ENC  64
#define TQ   128          // queries per block in fused kernel
#define LST  264          // padded LDS row stride (halves); 528B, 16B aligned
#define PI_F 3.14159265358979f

__device__ __forceinline__ int iclamp(int v, int lo, int hi) {
    return v < lo ? lo : (v > hi ? hi : v);
}

// nearest-sample index + relative coordinate for a given corner shift s
__device__ __forceinline__ void shift_rel(float c0, float c1, int s,
                                          int& iy, int& ix, float& r0, float& r1) {
    const float rx = 1.0f / Hh, ry = 1.0f / Ww;
    float vx = (s >> 1) ? 1.f : -1.f;
    float vy = (s & 1) ? 1.f : -1.f;
    float cy = fminf(fmaxf(c0 + vx * rx + 1e-6f, -1.f + 1e-6f), 1.f - 1e-6f);
    float cx = fminf(fmaxf(c1 + vy * ry + 1e-6f, -1.f + 1e-6f), 1.f - 1e-6f);
    float fiy = rintf((cy + 1.f) * (Hh * 0.5f) - 0.5f);
    float fix = rintf((cx + 1.f) * (Ww * 0.5f) - 0.5f);
    iy = iclamp((int)fiy, 0, Hh - 1);
    ix = iclamp((int)fix, 0, Ww - 1);
    float qy = -1.f + (2.f * iy + 1.f) / Hh;
    float qx = -1.f + (2.f * ix + 1.f) / Ww;
    r0 = (c0 - qy) * Hh;
    r1 = (c1 - qx) * Ww;
}

// ---------------------------------------------------------------------------
// Pack w0..w3 (each 256x256, row = out-neuron n, col = k) into WMMA B-fragment
// order: pw[(((l*16+jt)*8+kt)*32+lane)*16 + e] = w_l[n*256+k]
//   n = jt*16 + (lane&15),  k = kt*32 + (lane>>4)*16 + e
// ---------------------------------------------------------------------------
__global__ __launch_bounds__(256) void pack_w_kernel(
    const float* __restrict__ w0, const float* __restrict__ w1,
    const float* __restrict__ w2, const float* __restrict__ w3,
    _Float16* __restrict__ pw) {
    int idx = blockIdx.x * 256 + threadIdx.x;     // 0 .. 262143
    int l    = idx >> 16;
    int e    = idx & 15;
    int lane = (idx >> 4) & 31;
    int kt   = (idx >> 9) & 7;
    int jt   = (idx >> 12) & 15;
    int n = jt * 16 + (lane & 15);
    int k = kt * 32 + ((lane >> 4) << 4) + e;
    const float* w = (l == 0) ? w0 : (l == 1) ? w1 : (l == 2) ? w2 : w3;
    pw[idx] = (_Float16)w[n * 256 + k];
}

// ---------------------------------------------------------------------------
// enc conv: (2,3,64,64) -> (2,64,64,64), 3x3 pad 1
// ---------------------------------------------------------------------------
__global__ __launch_bounds__(256) void enc_conv_kernel(
    const float* __restrict__ inp, const float* __restrict__ w,
    const float* __restrict__ bias, float* __restrict__ feat) {
    int idx = blockIdx.x * 256 + threadIdx.x;     // 0 .. 524287
    int x = idx & 63, y = (idx >> 6) & 63, c = (idx >> 12) & 63, b = idx >> 18;
    float acc = bias[c];
    for (int ci = 0; ci < 3; ci++) {
        for (int dy = 0; dy < 3; dy++) {
            int yy = y + dy - 1;
            if (yy < 0 || yy >= Hh) continue;
            for (int dx = 0; dx < 3; dx++) {
                int xx = x + dx - 1;
                if (xx < 0 || xx >= Ww) continue;
                acc += inp[((b * 3 + ci) * Hh + yy) * Ww + xx] *
                       w[((c * 3 + ci) * 3 + dy) * 3 + dx];
            }
        }
    }
    feat[idx] = acc;
}

// ---------------------------------------------------------------------------
// 64->256 conv: feat (2,64,64,64) -> out (2,256,64,64), 3x3 pad 1
// ---------------------------------------------------------------------------
__global__ __launch_bounds__(256) void conv256_kernel(
    const float* __restrict__ feat, const float* __restrict__ w,
    const float* __restrict__ bias, float* __restrict__ out) {
    int idx = blockIdx.x * 256 + threadIdx.x;     // 0 .. 2097151
    int x = idx & 63, y = (idx >> 6) & 63, c = (idx >> 12) & 255, b = idx >> 20;
    float acc = bias[c];
    for (int dy = 0; dy < 3; dy++) {
        int yy = y + dy - 1;
        if (yy < 0 || yy >= Hh) continue;
        for (int dx = 0; dx < 3; dx++) {
            int xx = x + dx - 1;
            if (xx < 0 || xx >= Ww) continue;
            const float* fp = feat + ((size_t)b * ENC * Hh + yy) * Ww + xx;
            const float* wp = w + ((size_t)c * ENC * 3 + dy) * 3 + dx;
            for (int ci = 0; ci < ENC; ci++)
                acc += fp[(size_t)ci * Hh * Ww] * wp[(size_t)ci * 9];
        }
    }
    out[idx] = acc;
}

// ---------------------------------------------------------------------------
// Fused Fourier-feature + 4-layer 256x256 MLP (WMMA) + output head,
// looped over the 4 corner shifts with diagonal-area weighting.
// Block = 128 threads = 4 waves; each wave owns TWO private 16-row M-slabs
// (32 queries) so every B fragment feeds two WMMAs (halved weight traffic).
// All phases are wave-local on the same 32-row slab -> no barriers at all.
// ---------------------------------------------------------------------------
__global__ __launch_bounds__(128) void fused_mlp_kernel(
    const float* __restrict__ coord, const float* __restrict__ cell,
    const float* __restrict__ coef, const float* __restrict__ freq,
    const float* __restrict__ phw, const _Float16* __restrict__ pw,
    const float* __restrict__ b0, const float* __restrict__ b1,
    const float* __restrict__ b2, const float* __restrict__ b3,
    const float* __restrict__ wo, const float* __restrict__ bo,
    float* __restrict__ ret) {
    __shared__ _Float16 buf[2][TQ * LST];   // 2 x 128 x 264 f16 ping-pong

    const int tid  = threadIdx.x;           // 0..127, one query per thread
    const int lane = tid & 31;
    const int wave = tid >> 5;              // 0..3
    const int g    = blockIdx.x * TQ + tid;
    const int b    = g >> 16;               // Q == 65536
    const float c0 = coord[g * 2 + 0];
    const float c1 = coord[g * 2 + 1];
    const float rc0 = cell[g * 2 + 0] * (float)Hh;
    const float rc1 = cell[g * 2 + 1] * (float)Ww;

    float den = 0.f;
    for (int s = 0; s < 4; s++) {
        int iy, ix; float r0, r1;
        shift_rel(c0, c1, s, iy, ix, r0, r1);
        den += fabsf(r0 * r1) + 1e-9f;
    }

    float num0 = 0.f, num1 = 0.f, num2 = 0.f;

    for (int s = 0; s < 4; s++) {
        // ---- build x = coef * [cos|sin](pi*qf) into buf[0] row tid
        {
            int iy, ix; float r0, r1;
            shift_rel(c0, c1, s, iy, ix, r0, r1);
            const size_t pix = (size_t)iy * Ww + ix;
            const float* fq = freq + (size_t)b * HID * Hh * Ww + pix;
            const float* cf = coef + (size_t)b * HID * Hh * Ww + pix;
            _Float16* xrow = &buf[0][tid * LST];
            for (int k = 0; k < HID / 2; k++) {
                float f0 = fq[(size_t)(2 * k) * (Hh * Ww)];
                float f1 = fq[(size_t)(2 * k + 1) * (Hh * Ww)];
                float ph = rc0 * phw[2 * k] + rc1 * phw[2 * k + 1];
                float qf = PI_F * (f0 * r0 + f1 * r1 + ph);
                float sv, cv;
                __sincosf(qf, &sv, &cv);
                xrow[k]           = (_Float16)(cf[(size_t)k * (Hh * Ww)] * cv);
                xrow[k + HID / 2] = (_Float16)(cf[(size_t)(k + HID / 2) * (Hh * Ww)] * sv);
            }
        }
        // ---- 4 dense layers: LDS ping-pong, B streamed once per wave,
        //      two M-tiles per wave share each B fragment.
        for (int l = 0; l < 4; l++) {
            const int srcI = l & 1;              // x in buf0 -> h3 ends in buf0
            const _Float16* src = &buf[srcI][0];
            _Float16* dst = &buf[srcI ^ 1][0];
            const float* bl = (l == 0) ? b0 : (l == 1) ? b1 : (l == 2) ? b2 : b3;

            const int m0 = wave * 32 + (lane & 15);
            v16h a0[8], a1[8];
#pragma unroll
            for (int kt = 0; kt < 8; kt++) {
                const _Float16* p0 = src + m0 * LST + kt * 32 + ((lane >> 4) << 4);
                const _Float16* p1 = p0 + 16 * LST;
                v8h lo0 = *(const v8h*)p0, hi0 = *(const v8h*)(p0 + 8);
                v8h lo1 = *(const v8h*)p1, hi1 = *(const v8h*)(p1 + 8);
#pragma unroll
                for (int e = 0; e < 8; e++) {
                    a0[kt][e] = lo0[e]; a0[kt][8 + e] = hi0[e];
                    a1[kt][e] = lo1[e]; a1[kt][8 + e] = hi1[e];
                }
            }
            const int nlane = lane & 15;
            const int mrow0 = wave * 32 + ((lane >> 4) << 3);
            for (int jt = 0; jt < 16; jt++) {
                float bv = bl[jt * 16 + nlane];
                v8f acc0 = {bv, bv, bv, bv, bv, bv, bv, bv};
                v8f acc1 = acc0;
                const _Float16* bbase = pw + (size_t)(((l * 16 + jt) * 8) * 32 + lane) * 16;
#pragma unroll
                for (int kt = 0; kt < 8; kt++) {
                    const _Float16* qp = bbase + kt * 512;
                    v8h blo = *(const v8h*)qp;
                    v8h bhi = *(const v8h*)(qp + 8);
                    v16h bf;
#pragma unroll
                    for (int e = 0; e < 8; e++) { bf[e] = blo[e]; bf[8 + e] = bhi[e]; }
                    acc0 = __builtin_amdgcn_wmma_f32_16x16x32_f16(
                        false, a0[kt], false, bf, (short)0, acc0, false, false);
                    acc1 = __builtin_amdgcn_wmma_f32_16x16x32_f16(
                        false, a1[kt], false, bf, (short)0, acc1, false, false);
                }
                int n = jt * 16 + nlane;
#pragma unroll
                for (int r = 0; r < 8; r++) {
                    float v0 = acc0[r]; v0 = v0 > 0.f ? v0 : 0.f;      // ReLU
                    dst[(mrow0 + r) * LST + n] = (_Float16)v0;
                    float v1 = acc1[r]; v1 = v1 > 0.f ? v1 : 0.f;
                    dst[(mrow0 + 16 + r) * LST + n] = (_Float16)v1;
                }
            }
        }
        // ---- output head (N=3) + diagonal-area-weighted accumulation
        {
            float p0 = bo[0], p1 = bo[1], p2 = bo[2];
            const _Float16* hr = &buf[0][tid * LST];
            for (int c = 0; c < HID; c += 8) {
                v8h hv8 = *(const v8h*)(hr + c);
#pragma unroll
                for (int e = 0; e < 8; e++) {
                    float hv = (float)hv8[e];
                    p0 += hv * wo[c + e];
                    p1 += hv * wo[HID + c + e];
                    p2 += hv * wo[2 * HID + c + e];
                }
            }
            int iy, ix; float r0, r1;
            shift_rel(c0, c1, 3 - s, iy, ix, r0, r1);
            float wgt = fabsf(r0 * r1) + 1e-9f;
            num0 += p0 * wgt; num1 += p1 * wgt; num2 += p2 * wgt;
        }
    }
    float inv = 1.f / den;
    ret[g * 3 + 0] = num0 * inv;
    ret[g * 3 + 1] = num1 * inv;
    ret[g * 3 + 2] = num2 * inv;
}

// ---------------------------------------------------------------------------
// Residual bilinear sample + per-block BN partial sums (sum, sum^2 per ch)
// ---------------------------------------------------------------------------
__global__ __launch_bounds__(256) void finalize_kernel(
    const float* __restrict__ inp, const float* __restrict__ coord,
    float* __restrict__ ret, float* __restrict__ partials) {
    __shared__ float red[256];
    int tid = threadIdx.x;
    int g = blockIdx.x * 256 + tid;               // < 131072
    int b = g >> 16;
    float c0 = coord[g * 2 + 0], c1 = coord[g * 2 + 1];
    float fy = (c0 + 1.f) * (Hh * 0.5f) - 0.5f;
    float fx = (c1 + 1.f) * (Ww * 0.5f) - 0.5f;
    float y0 = floorf(fy), x0 = floorf(fx);
    float wy = fy - y0, wx = fx - x0;
    int y0c = iclamp((int)y0, 0, Hh - 1);
    int y1c = iclamp((int)y0 + 1, 0, Hh - 1);
    int x0c = iclamp((int)x0, 0, Ww - 1);
    int x1c = iclamp((int)x0 + 1, 0, Ww - 1);
    float vals[6];
    for (int c = 0; c < 3; c++) {
        const float* ip = inp + (size_t)(b * 3 + c) * Hh * Ww;
        float v00 = ip[y0c * Ww + x0c], v01 = ip[y0c * Ww + x1c];
        float v10 = ip[y1c * Ww + x0c], v11 = ip[y1c * Ww + x1c];
        float res = v00 * (1 - wy) * (1 - wx) + v01 * (1 - wy) * wx +
                    v10 * wy * (1 - wx) + v11 * wy * wx;
        float v = ret[g * 3 + c] + res;
        ret[g * 3 + c] = v;
        vals[c] = v;
        vals[3 + c] = v * v;
    }
    for (int k = 0; k < 6; k++) {
        red[tid] = vals[k];
        __syncthreads();
        for (int st = 128; st > 0; st >>= 1) {
            if (tid < st) red[tid] += red[tid + st];
            __syncthreads();
        }
        if (tid == 0) partials[blockIdx.x * 6 + k] = red[0];
        __syncthreads();
    }
}

__global__ __launch_bounds__(256) void reduce_kernel(
    const float* __restrict__ partials, float* __restrict__ stats) {
    __shared__ float red[256];
    __shared__ float tot[6];
    int tid = threadIdx.x;
    for (int k = 0; k < 6; k++) {
        float a = 0.f;
        for (int i = tid; i < 512; i += 256) a += partials[i * 6 + k];
        red[tid] = a;
        __syncthreads();
        for (int st = 128; st > 0; st >>= 1) {
            if (tid < st) red[tid] += red[tid + st];
            __syncthreads();
        }
        if (tid == 0) tot[k] = red[0];
        __syncthreads();
    }
    if (tid == 0) {
        const float n = (float)(Bn * Qq);
        for (int c = 0; c < 3; c++) {
            float mean = tot[c] / n;
            float var = tot[3 + c] / n - mean * mean;
            stats[c] = mean;
            stats[3 + c] = rsqrtf(var + 1e-5f);
        }
    }
}

__global__ __launch_bounds__(256) void bn_kernel(
    const float* __restrict__ ret, const float* __restrict__ stats,
    const float* __restrict__ gamma, const float* __restrict__ beta,
    float* __restrict__ out) {
    int idx = blockIdx.x * 256 + threadIdx.x;     // < 393216
    int c = idx % 3;
    out[idx] = (ret[idx] - stats[c]) * stats[3 + c] * gamma[c] + beta[c];
}

// ---------------------------------------------------------------------------
extern "C" void kernel_launch(void* const* d_in, const int* in_sizes, int n_in,
                              void* d_out, int out_size, void* d_ws, size_t ws_size,
                              hipStream_t stream) {
    (void)in_sizes; (void)n_in; (void)out_size; (void)ws_size;
    const float* inp    = (const float*)d_in[0];
    const float* coord  = (const float*)d_in[1];
    const float* cell   = (const float*)d_in[2];
    const float* enc_w  = (const float*)d_in[3];
    const float* enc_b  = (const float*)d_in[4];
    const float* coef_w = (const float*)d_in[5];
    const float* coef_b = (const float*)d_in[6];
    const float* freq_w = (const float*)d_in[7];
    const float* freq_b = (const float*)d_in[8];
    const float* phw    = (const float*)d_in[9];
    const float* w0 = (const float*)d_in[10]; const float* b0 = (const float*)d_in[11];
    const float* w1 = (const float*)d_in[12]; const float* b1 = (const float*)d_in[13];
    const float* w2 = (const float*)d_in[14]; const float* b2 = (const float*)d_in[15];
    const float* w3 = (const float*)d_in[16]; const float* b3 = (const float*)d_in[17];
    const float* wo = (const float*)d_in[18]; const float* bo = (const float*)d_in[19];
    const float* gamma = (const float*)d_in[20];
    const float* beta  = (const float*)d_in[21];

    char* ws = (char*)d_ws;
    float*     feat     = (float*)(ws);                        //  2 MB
    float*     coef     = (float*)(ws + 2097152);              //  8 MB
    float*     freq     = (float*)(ws + 10485760);             //  8 MB
    _Float16*  pw       = (_Float16*)(ws + 18874368);          // 512 KB
    float*     ret      = (float*)(ws + 19398656);             // 1.5 MB
    float*     partials = (float*)(ws + 20971520);             // 12 KB
    float*     stats    = (float*)(ws + 20983808);             // 24 B

    pack_w_kernel<<<1024, 256, 0, stream>>>(w0, w1, w2, w3, pw);
    enc_conv_kernel<<<2048, 256, 0, stream>>>(inp, enc_w, enc_b, feat);
    conv256_kernel<<<8192, 256, 0, stream>>>(feat, coef_w, coef_b, coef);
    conv256_kernel<<<8192, 256, 0, stream>>>(feat, freq_w, freq_b, freq);
    fused_mlp_kernel<<<1024, 128, 0, stream>>>(coord, cell, coef, freq, phw, pw,
                                               b0, b1, b2, b3, wo, bo, ret);
    finalize_kernel<<<512, 256, 0, stream>>>(inp, coord, ret, partials);
    reduce_kernel<<<1, 256, 0, stream>>>(partials, stats);
    bn_kernel<<<1536, 256, 0, stream>>>(ret, stats, gamma, beta, (float*)d_out);
}
